// GAT_15762529976639
// MI455X (gfx1250) — matrix operand order repaired and checked by reading, not compiled, not run on hardware
//
#include <hip/hip_runtime.h>

// Problem constants (match reference)
#define B_    4
#define N_    4096
#define C_    64
#define ALPHA 0.2f
#define THRESH 0.5f
#define NEGINF (-9.0e15f)

typedef __attribute__((ext_vector_type(16))) _Float16 v16h;
typedef __attribute__((ext_vector_type(8)))  _Float16 v8h;
typedef __attribute__((ext_vector_type(8)))  float    v8f;
typedef __attribute__((ext_vector_type(4)))  float    v4f;

// ---------------------------------------------------------------------------
// Kernel 0: s1[b][n] = h[b][n][:]·a[0:C], s2[b][n] = h[b][n][:]·a[C:2C],
// and hT16[b][c][n] = (f16)h[b][n][c]  (transposed f16 copy of h so the WMMA
// B-matrix per-lane data is two contiguous 16B runs along j).
// ---------------------------------------------------------------------------
__global__ __launch_bounds__(256)
void gat_prep(const float* __restrict__ h, const float* __restrict__ a,
              float* __restrict__ s1, float* __restrict__ s2,
              _Float16* __restrict__ hT) {
    int idx = blockIdx.x * blockDim.x + threadIdx.x;   // 0 .. B*N-1
    int b = idx >> 12;          // / N_
    int n = idx & (N_ - 1);
    const float* hp = h + (size_t)idx * C_;
    float acc1 = 0.f, acc2 = 0.f;
#pragma unroll
    for (int c = 0; c < C_; ++c) {
        float v = hp[c];
        acc1 += v * a[c];
        acc2 += v * a[C_ + c];
        hT[((size_t)b * C_ + c) * N_ + n] = (_Float16)v;   // coalesced across lanes (n)
    }
    s1[idx] = acc1;
    s2[idx] = acc2;
}

// ---------------------------------------------------------------------------
// Kernel 1: fused masked-softmax attention * h, flash-attention style.
// One wave32 handles a 16-row i-tile (16x64 f32 output), looping j in chunks
// of 32 (WMMA K). Per-lane e/p values are laid out directly in the CDNA5
// 16-bit A-matrix layout:
//   row m = lane&15, hi = lane>>4
//   half t<8  ->  K = 8*hi + t
//   half t>=8 ->  K = 16 + 8*hi + (t-8)
// C/D f32 layout: col n = lane&15, VGPR r -> row m = r + 8*hi.
// ---------------------------------------------------------------------------
__global__ __launch_bounds__(256, 1)
void gat_attn(const float* __restrict__ adj, const float* __restrict__ s1,
              const float* __restrict__ s2, const _Float16* __restrict__ hT,
              float* __restrict__ out) {
    const int lane = threadIdx.x & 31;
    const int wave = threadIdx.x >> 5;
    const int b    = blockIdx.y;
    const int i0   = (blockIdx.x * 8 + wave) * 16;
    const int m    = lane & 15;   // tile row owned by this lane (A/stats layout)
    const int hi   = lane >> 4;   // K-group select

    const float     my_s1  = s1[b * N_ + i0 + m];
    const float*    adjRow = adj + (size_t)(i0 + m) * N_;
    const float*    s2b    = s2 + b * N_;
    const _Float16* hTb    = hT + (size_t)b * C_ * N_;

    float rowMax = -1.0e30f;
    float rowSum = 0.0f;
    v8f acc[4] = {v8f{}, v8f{}, v8f{}, v8f{}};

    for (int j0 = 0; j0 < N_; j0 += 32) {
        const int kbase0 = j0 + 8 * hi;        // K block for halves t=0..7
        const int kbase1 = j0 + 16 + 8 * hi;   // K block for halves t=8..15

        if (j0 + 32 < N_) {                    // prefetch next adj chunk
            __builtin_prefetch(adjRow + kbase0 + 32, 0, 1);
        }

        // 32B-aligned vector loads of adj and s2 in A-matrix half order
        v4f av[4], qv[4];
        av[0] = *(const v4f*)(adjRow + kbase0);
        av[1] = *(const v4f*)(adjRow + kbase0 + 4);
        av[2] = *(const v4f*)(adjRow + kbase1);
        av[3] = *(const v4f*)(adjRow + kbase1 + 4);
        qv[0] = *(const v4f*)(s2b + kbase0);
        qv[1] = *(const v4f*)(s2b + kbase0 + 4);
        qv[2] = *(const v4f*)(s2b + kbase1);
        qv[3] = *(const v4f*)(s2b + kbase1 + 4);

        // e = leakyrelu(s1_i + s2_j), masked by adj > 0.5
        float p[16];
        float chunkMax = NEGINF;
#pragma unroll
        for (int t = 0; t < 16; ++t) {
            float e = my_s1 + qv[t >> 2][t & 3];
            e = e > 0.f ? e : ALPHA * e;
            e = (av[t >> 2][t & 3] > THRESH) ? e : NEGINF;
            p[t] = e;
            chunkMax = fmaxf(chunkMax, e);
        }
        // combine with the partner lane holding the other half of this row's K
        chunkMax = fmaxf(chunkMax, __shfl_xor(chunkMax, 16, 32));
        float newMax = fmaxf(rowMax, chunkMax);
        float scale  = __expf(rowMax - newMax);

        float lsum = 0.f;
#pragma unroll
        for (int t = 0; t < 16; ++t) {
            float pe = __expf(p[t] - newMax);
            p[t] = pe;
            lsum += pe;
        }
        lsum  += __shfl_xor(lsum, 16, 32);
        rowSum = rowSum * scale + lsum;
        rowMax = newMax;

        // rescale accumulators: D VGPR r holds row (r + 8*hi); scale lives in lane m
        float rs[8];
#pragma unroll
        for (int r = 0; r < 8; ++r) rs[r] = __shfl(scale, r + 8 * hi, 32);
#pragma unroll
        for (int r = 0; r < 8; ++r) {
            acc[0][r] *= rs[r]; acc[1][r] *= rs[r];
            acc[2][r] *= rs[r]; acc[3][r] *= rs[r];
        }

        // pack P into the f16 A matrix (layout already matches half ordering)
        v16h aMat;
#pragma unroll
        for (int t = 0; t < 16; ++t) aMat[t] = (_Float16)p[t];

        // 4 WMMA tiles cover C=64: B per-lane = hT[b][c0+n][j0+8*hi .. +8) and [j0+16+8*hi .. +8)
#pragma unroll
        for (int ct = 0; ct < 4; ++ct) {
            const _Float16* src = hTb + (size_t)(ct * 16 + m) * N_ + j0 + hi * 8;
            v8h b0 = *(const v8h*)(src);
            v8h b1 = *(const v8h*)(src + 16);
            v16h bMat;
#pragma unroll
            for (int t = 0; t < 8; ++t) { bMat[t] = b0[t]; bMat[t + 8] = b1[t]; }
            acc[ct] = __builtin_amdgcn_wmma_f32_16x16x32_f16(
                false, aMat, false, bMat, (short)0, acc[ct], false, false);
        }
    }

    // finalize: divide by softmax denom, apply elu, store 16x64 f32 tile
    float invL = 1.0f / rowSum;                 // per row m (lanes m and m+16 agree)
    float il[8];
#pragma unroll
    for (int r = 0; r < 8; ++r) il[r] = __shfl(invL, r + 8 * hi, 32);

#pragma unroll
    for (int ct = 0; ct < 4; ++ct) {
#pragma unroll
        for (int r = 0; r < 8; ++r) {
            float v = acc[ct][r] * il[r];
            v = v > 0.f ? v : (__expf(v) - 1.0f);   // elu
            int row = r + 8 * hi;
            out[((size_t)(b * N_ + i0 + row)) * C_ + ct * 16 + m] = v;
        }
    }
}

// ---------------------------------------------------------------------------
extern "C" void kernel_launch(void* const* d_in, const int* in_sizes, int n_in,
                              void* d_out, int out_size, void* d_ws, size_t ws_size,
                              hipStream_t stream) {
    const float* h   = (const float*)d_in[0];   // (B,N,C) f32
    const float* adj = (const float*)d_in[1];   // (N,N)   f32
    const float* a   = (const float*)d_in[2];   // (2C,1)  f32
    float* out = (float*)d_out;                 // (B,N,C) f32

    // workspace layout: s1 | s2 | hT16   (~2.1 MB total)
    float*    s1 = (float*)d_ws;
    float*    s2 = s1 + (size_t)B_ * N_;
    _Float16* hT = (_Float16*)(s2 + (size_t)B_ * N_);

    gat_prep<<<(B_ * N_) / 256, 256, 0, stream>>>(h, a, s1, s2, hT);
    gat_attn<<<dim3(N_ / 128, B_), 256, 0, stream>>>(adj, s1, s2, hT, out);
}
